// DeepISTA_63960652972575
// MI455X (gfx1250) — matrix operand order repaired and checked by reading, not compiled, not run on hardware
//
#include <hip/hip_runtime.h>
#include <hip/hip_bf16.h>
#include <hip/hip_fp16.h>

// ---------------------------------------------------------------------------
// FISTA / DeepISTA on MI455X (gfx1250)
//   iteration rewritten with the Gram matrix:  z' = soft(z - (step*G)z + step*D^T y)
//   G = D^T D (512x512, f16, pre-scaled, pre-swizzled to WMMA A-operand layout)
//   per-column state stays resident in LDS for all 100 iterations.
//   kb loop kept rolled: steady state = 4 ds_load_b128 + 8 global_load_b128 +
//   8 v_wmma_f32_16x16x32_f16, 8 live accumulators, all A tiles in distinct
//   registers so load latency overlaps WMMA issue.
// ---------------------------------------------------------------------------

typedef __attribute__((ext_vector_type(16))) _Float16 v16h;
typedef __attribute__((ext_vector_type(8)))  float    v8f;

#define DIMY    256
#define DIMX    512
#define NS      16384
#define NITER   100
#define STEP    0.125f            // 1/LIPSCHITZ
#define THR     (0.125f * 0.1f)   // step * lambda
#define COLS_WG 32                // sample columns per workgroup

// ---- swizzled-offset helpers (16-bit WMMA operand layouts, ISA 7.12.2) ----

// A operand 16x32 f16: lanes 0-15 -> M, lanes 16-31 -> M (other K group)
//   lane = ((Kl>>3)&1)*16 + Ml ; half index h = (Kl>>4)*8 + (Kl&7)
__device__ __forceinline__ int a_swz(int Ml, int Kl) {
    int lane = (((Kl >> 3) & 1) << 4) + Ml;
    int h    = ((Kl >> 4) << 3) + (Kl & 7);
    return lane * 16 + h;
}

// B operand 32x16 f16 (same K order as A): lane = ((K>>3)&1)*16 + Nl,
// within lane: h = (K>>4)*8 + (K&7)
__device__ __forceinline__ int z_idx(int row, int c) {
    int kb = row >> 5;          // K-tile (32 wide)
    int Kl = row & 31;
    int n  = c >> 4;            // N-tile (16 wide)
    int Nl = c & 15;
    return ((kb << 1) + n) * 512 + ((((Kl >> 3) & 1) << 4) + Nl) * 16 + ((Kl >> 4) << 3) + (Kl & 7);
}

// ---------------------------------------------------------------------------
// Kernel 1a: Gs = step * D^T D   (f16, A-operand swizzled tiles of 16Mx32K)
// ---------------------------------------------------------------------------
__global__ void deepista_gram(const float* __restrict__ D,
                              _Float16* __restrict__ Gs) {
    int t = blockIdx.x * 256 + threadIdx.x;   // 512*512 threads
    int i = t >> 9;       // output row (M)
    int j = t & 511;      // output col (K)
    float s = 0.0f;
    #pragma unroll 8
    for (int k = 0; k < DIMY; ++k)
        s += D[k * DIMX + i] * D[k * DIMX + j];
    s *= STEP;
    int mT = i >> 4, Ml = i & 15;
    int kb = j >> 5, Kl = j & 31;
    Gs[(mT * 16 + kb) * 512 + a_swz(Ml, Kl)] = (_Float16)s;
}

// ---------------------------------------------------------------------------
// Kernel 1b: cs = step * D^T y   (f16, row-major [512][16384])
// ---------------------------------------------------------------------------
__global__ void deepista_dty(const float* __restrict__ D,
                             const float* __restrict__ y,
                             _Float16* __restrict__ cs) {
    int colBase = (blockIdx.x & 63) * 256;     // 64 column groups
    int iBase   = (blockIdx.x >> 6) * 16;      // 32 row groups of 16
    int col     = colBase + threadIdx.x;
    float acc[16];
    #pragma unroll
    for (int ii = 0; ii < 16; ++ii) acc[ii] = 0.0f;
    for (int k = 0; k < DIMY; ++k) {
        float yv = y[k * NS + col];
        #pragma unroll
        for (int ii = 0; ii < 16; ++ii)
            acc[ii] += D[k * DIMX + iBase + ii] * yv;   // uniform D -> SGPR
    }
    #pragma unroll
    for (int ii = 0; ii < 16; ++ii)
        cs[(size_t)(iBase + ii) * NS + col] = (_Float16)(STEP * acc[ii]);
}

// ---------------------------------------------------------------------------
// Kernel 2: 100 resident FISTA iterations per 32-column block.
//   LDS (128 KB dynamic): z double buffer (2 x 32 KB, B-operand swizzled),
//   out_old (32 KB), cs block (32 KB).  2 workgroups / WGP.
// ---------------------------------------------------------------------------
__global__ void deepista_fista(const _Float16* __restrict__ Gs,
                               const _Float16* __restrict__ cs,
                               float* __restrict__ out) {
    extern __shared__ _Float16 smem[];
    _Float16* zb0  = smem;               // 16384 halves
    _Float16* zb1  = smem + 16384;       // 16384 halves
    _Float16* oldB = smem + 2 * 16384;   // 16384 halves  [row*32 + c]
    _Float16* csB  = smem + 3 * 16384;   // 16384 halves  [row*32 + c]

    const int tid     = threadIdx.x;
    const int lane    = tid & 31;
    const int wave    = tid >> 5;            // 8 waves
    const int colBase = blockIdx.x * COLS_WG;

    // init: z = 0, out_old = 0, stage cs block
    for (int idx = tid; idx < 16384; idx += 256) {
        zb0[idx]  = (_Float16)0.0f;
        oldB[idx] = (_Float16)0.0f;
        int row = idx >> 5, c = idx & 31;
        csB[idx] = cs[(size_t)row * NS + colBase + c];
    }
    __syncthreads();

    float t_old = 1.0f;
    int   cur   = 0;

    #pragma unroll 1
    for (int it = 0; it < NITER; ++it) {
        float t_new = 0.5f * (1.0f + sqrtf(1.0f + 4.0f * t_old * t_old));
        float mom   = (t_old - 1.0f) / t_new;
        const _Float16* zc = cur ? zb1 : zb0;
        _Float16*       zn = cur ? zb0 : zb1;

        // each wave owns 4 M-tiles (rows m*16..m*16+15) x both N-tiles;
        // all 8 accumulators live across the K loop -> each B tile read once.
        // K loop stays ROLLED; A tiles go to 4 distinct register sets so all
        // loads can issue before the WMMA chain consumes them.
        v8f acc[4][2] = {};
        #pragma unroll 1
        for (int kb = 0; kb < 16; ++kb) {
            const _Float16* gp = Gs + ((wave * 4) * 16 + kb) * 512 + lane * 16;
            v16h a0 = *(const v16h*)(gp);
            v16h a1 = *(const v16h*)(gp + 16 * 512);
            v16h a2 = *(const v16h*)(gp + 32 * 512);
            v16h a3 = *(const v16h*)(gp + 48 * 512);
            v16h b0 = *(const v16h*)(zc + (((kb << 1) + 0) * 512 + lane * 16));
            v16h b1 = *(const v16h*)(zc + (((kb << 1) + 1) * 512 + lane * 16));
            acc[0][0] = __builtin_amdgcn_wmma_f32_16x16x32_f16(
                    false, a0, false, b0, (short)0, acc[0][0], false, false);
            acc[0][1] = __builtin_amdgcn_wmma_f32_16x16x32_f16(
                    false, a0, false, b1, (short)0, acc[0][1], false, false);
            acc[1][0] = __builtin_amdgcn_wmma_f32_16x16x32_f16(
                    false, a1, false, b0, (short)0, acc[1][0], false, false);
            acc[1][1] = __builtin_amdgcn_wmma_f32_16x16x32_f16(
                    false, a1, false, b1, (short)0, acc[1][1], false, false);
            acc[2][0] = __builtin_amdgcn_wmma_f32_16x16x32_f16(
                    false, a2, false, b0, (short)0, acc[2][0], false, false);
            acc[2][1] = __builtin_amdgcn_wmma_f32_16x16x32_f16(
                    false, a2, false, b1, (short)0, acc[2][1], false, false);
            acc[3][0] = __builtin_amdgcn_wmma_f32_16x16x32_f16(
                    false, a3, false, b0, (short)0, acc[3][0], false, false);
            acc[3][1] = __builtin_amdgcn_wmma_f32_16x16x32_f16(
                    false, a3, false, b1, (short)0, acc[3][1], false, false);
        }

        // fused elementwise: C-layout coords -> (row, col)
        const int cHalf = lane & 15;
        #pragma unroll
        for (int mi = 0; mi < 4; ++mi) {
            int rBase = (wave * 4 + mi) * 16 + ((lane >> 4) << 3);
            #pragma unroll
            for (int n = 0; n < 2; ++n) {
                int c = n * 16 + cHalf;
                #pragma unroll
                for (int v = 0; v < 8; ++v) {
                    int row  = rBase + v;
                    float wv = acc[mi][n][v];
                    int eidx = row * 32 + c;
                    float zf = (float)zc[z_idx(row, c)];
                    float p  = zf - wv + (float)csB[eidx];       // gradient step
                    float ap = fabsf(p) - THR;                   // soft threshold
                    ap = ap > 0.0f ? ap : 0.0f;
                    float onew = copysignf(ap, p);
                    float oold = (float)oldB[eidx];
                    float znew = onew + mom * (onew - oold);     // FISTA momentum
                    oldB[eidx]        = (_Float16)onew;
                    zn[z_idx(row, c)] = (_Float16)znew;
                    if (it == NITER - 1)
                        out[(size_t)row * NS + colBase + c] = znew;
                }
            }
        }
        __syncthreads();
        cur ^= 1;
        t_old = t_new;
    }
}

// ---------------------------------------------------------------------------
extern "C" void kernel_launch(void* const* d_in, const int* in_sizes, int n_in,
                              void* d_out, int out_size, void* d_ws, size_t ws_size,
                              hipStream_t stream) {
    const float* y = (const float*)d_in[0];   // (256, 16384)
    const float* D = (const float*)d_in[1];   // (256, 512)
    float* out = (float*)d_out;               // (512, 16384)

    _Float16* Gs = (_Float16*)d_ws;                              // 512 KB
    _Float16* cs = (_Float16*)((char*)d_ws + 512 * 1024);        // 16 MB

    (void)in_sizes; (void)n_in; (void)out_size; (void)ws_size;

    // allow 128 KB dynamic LDS (gfx1250 WGP has 320 KB)
    hipFuncSetAttribute((const void*)deepista_fista,
                        hipFuncAttributeMaxDynamicSharedMemorySize, 131072);

    deepista_gram <<< 1024, 256, 0, stream >>> (D, Gs);
    deepista_dty  <<< 2048, 256, 0, stream >>> (D, y, cs);
    deepista_fista<<< NS / COLS_WG, 256, 131072, stream >>> (Gs, cs, out);
}
